// SimpleAttention_32066225832529
// MI455X (gfx1250) — matrix operand order repaired and checked by reading, not compiled, not run on hardware
//
#include <hip/hip_runtime.h>

// ---------------------------------------------------------------------------
// Problem constants (from reference): B=2, T=2048, C=1024, H=16, HS=64
// ---------------------------------------------------------------------------
constexpr int Bc  = 2;
constexpr int Tc  = 2048;
constexpr int Cc  = 1024;
constexpr int Hc  = 16;
constexpr int HSc = 64;
constexpr int Mc  = Bc * Tc;         // 4096 rows for all GEMMs
constexpr int KT  = Tc / 32;         // 64 key tiles of 32
constexpr float ATT_SCALE = 0.125f;  // 1/sqrt(64)

typedef __bf16 bf16_t;
typedef bf16_t v8bf  __attribute__((ext_vector_type(8)));
typedef bf16_t v16bf __attribute__((ext_vector_type(16)));
typedef float  v8f   __attribute__((ext_vector_type(8)));

// round-to-nearest-even f32 -> bf16
__device__ inline bf16_t to_bf16(float f) {
  unsigned u = __builtin_bit_cast(unsigned, f);
  u += 0x7FFFu + ((u >> 16) & 1u);
  unsigned short h = (unsigned short)(u >> 16);
  return __builtin_bit_cast(bf16_t, h);
}

__device__ inline v8f wmma_bf16f32(v16bf a, v16bf b, v8f c) {
  // D = A(16x32 bf16) * B(32x16 bf16) + C(16x16 f32)
  return __builtin_amdgcn_wmma_f32_16x16x32_bf16(
      /*neg_a=*/false, a, /*neg_b=*/false, b,
      /*c_mod=*/(short)0, c, /*reuse_a=*/false, /*reuse_b=*/false);
}

// A-matrix fragment (16x32 bf16, row-major source, ld elements per row).
// ISA 7.12.2: lanes 0-15 row m, K = 0..7 (v0-3) and 16..23 (v4-7);
// lanes 16-31 row m, K = 8..15 and 24..31.  => two contiguous 16B loads.
__device__ inline v16bf load_a16x32(const bf16_t* Atile, int ld, int lane) {
  const int m = lane & 15, hi = lane >> 4;
  const bf16_t* p = Atile + (size_t)m * ld + 8 * hi;
  v8bf lo  = *(const v8bf*)(p);
  v8bf hi8 = *(const v8bf*)(p + 16);
  v16bf r;
  #pragma unroll
  for (int i = 0; i < 8; ++i) { r[i] = lo[i]; r[i + 8] = hi8[i]; }
  return r;
}

// B-matrix fragment (32x16 bf16) where B[k][n] = Wsrc[n*ld + k]
// (i.e. B = Wsrc^T, Wsrc row-major).  Lanes 0-15 col n hold K=0..7 and
// 16..23, lanes 16-31 hold K=8..15 and 24..31.  => two contiguous 16B loads.
__device__ inline v16bf load_b32x16T(const bf16_t* Wsrc, int ld, int lane) {
  const int n = lane & 15, hi = lane >> 4;
  const bf16_t* p = Wsrc + (size_t)n * ld + 8 * hi;
  v8bf lo  = *(const v8bf*)(p);
  v8bf hi8 = *(const v8bf*)(p + 16);
  v16bf r;
  #pragma unroll
  for (int i = 0; i < 8; ++i) { r[i] = lo[i]; r[i + 8] = hi8[i]; }
  return r;
}

// ---------------------------------------------------------------------------
// Kernel 1: f32 -> bf16 conversion (grid-stride)
// ---------------------------------------------------------------------------
__global__ void cvt_f32_bf16(const float* __restrict__ in,
                             bf16_t* __restrict__ out, int n) {
  int i = blockIdx.x * blockDim.x + threadIdx.x;
  int stride = gridDim.x * blockDim.x;
  for (; i < n; i += stride) out[i] = to_bf16(in[i]);
}

// ---------------------------------------------------------------------------
// Kernel 2/5: GEMM  out = A @ W^T   (A: [Mc,Cc] bf16, W: [Cc,Cc] bf16, both
// row-major).  Each wave owns a 32x64 output tile; K-loop in steps of 32.
// mode 0: bf16 out [B,H,T,HS]   (Q, K)
// mode 1: bf16 out [B,H,HS,T]   (V transposed for P@V B-fragments)
// mode 2: f32  out [Mc,Cc]      (final projection -> d_out)
// ---------------------------------------------------------------------------
__global__ __launch_bounds__(128) void gemm_bf16_wmma(
    const bf16_t* __restrict__ A, const bf16_t* __restrict__ W,
    void* __restrict__ out, int mode) {
  const int lane = threadIdx.x & 31;
  const int wave = blockIdx.x * (blockDim.x >> 5) + (threadIdx.x >> 5);
  const int tilesN = Cc / 64;                 // 16
  const int m0 = (wave / tilesN) * 32;
  const int n0 = (wave % tilesN) * 64;
  if (m0 >= Mc) return;                       // wave-uniform

  v8f acc[2][4];
  #pragma unroll
  for (int i = 0; i < 2; ++i)
    #pragma unroll
    for (int j = 0; j < 4; ++j)
      #pragma unroll
      for (int r = 0; r < 8; ++r) acc[i][j][r] = 0.f;

  for (int k0 = 0; k0 < Cc; k0 += 32) {
    v16bf a0 = load_a16x32(A + (size_t)m0 * Cc + k0, Cc, lane);
    v16bf a1 = load_a16x32(A + (size_t)(m0 + 16) * Cc + k0, Cc, lane);
    #pragma unroll
    for (int f = 0; f < 4; ++f) {
      v16bf bf = load_b32x16T(W + (size_t)(n0 + 16 * f) * Cc + k0, Cc, lane);
      acc[0][f] = wmma_bf16f32(a0, bf, acc[0][f]);
      acc[1][f] = wmma_bf16f32(a1, bf, acc[1][f]);
    }
  }

  // C/D layout: VGPR r -> row r (lanes 0-15) / row r+8 (lanes 16-31);
  // column = lane & 15.
  const int cn = lane & 15, hi = lane >> 4;
  #pragma unroll
  for (int mi = 0; mi < 2; ++mi)
    #pragma unroll
    for (int f = 0; f < 4; ++f)
      #pragma unroll
      for (int r = 0; r < 8; ++r) {
        const int mg = m0 + 16 * mi + r + 8 * hi;
        const int ng = n0 + 16 * f + cn;
        const float v = acc[mi][f][r];
        if (mode == 2) {
          ((float*)out)[(size_t)mg * Cc + ng] = v;
        } else {
          const int b = mg >> 11, t = mg & (Tc - 1);   // T = 2048
          const int h = ng >> 6,  d = ng & (HSc - 1);
          const bf16_t bv = to_bf16(v);
          if (mode == 0)
            ((bf16_t*)out)[(((size_t)(b * Hc + h)) * Tc + t) * HSc + d] = bv;
          else
            ((bf16_t*)out)[(((size_t)(b * Hc + h)) * HSc + d) * Tc + t] = bv;
        }
      }
}

// ---------------------------------------------------------------------------
// Kernel 3: per-(b,h) prefix sums of V over keys at 32-key granularity.
// PS[bh][kt][d] = sum_{j < 32*kt} V[j][d],  kt in 0..KT (inclusive) -> f32.
// Used to fold the entire masked tail of the softmax analytically.
// ---------------------------------------------------------------------------
__global__ __launch_bounds__(64) void vprefix32(const bf16_t* __restrict__ Vt,
                                                float* __restrict__ PS) {
  const int bh = blockIdx.x;           // B*H blocks
  const int d  = threadIdx.x;          // 0..63
  const bf16_t* vrow = Vt + (size_t)bh * HSc * Tc + (size_t)d * Tc;
  float* ps = PS + (size_t)bh * (KT + 1) * HSc;
  float acc = 0.f;
  ps[d] = 0.f;
  for (int kt = 0; kt < KT; ++kt) {
    #pragma unroll 8
    for (int j = 0; j < 32; ++j) acc += (float)vrow[kt * 32 + j];
    ps[(size_t)(kt + 1) * HSc + d] = acc;
  }
}

// ---------------------------------------------------------------------------
// Kernel 4: attention.  One wave per (b, h, 16-query tile).
//
// Reference quirk: strict-upper-triangle scores are set to 0 (NOT -inf) and
// the softmax runs over the FULL row, so masked columns carry weight
// exp(-m)/Z and contribute exp(-m)*V[j,:] to y.  Key observations:
//  * running max m starts at 0 and never decreases, and every tile past the
//    diagonal has all-zero scores => m is FINAL after the causal prefix and
//    the rescale factor for the tail is exactly 1.
//  * therefore the whole masked tail collapses to
//        l   += (T - 32*steps) * exp(-m)
//        O   += exp(-m) * (PS[KT] - PS[steps])          (per d)
//    which we add analytically from the precomputed V prefix sums.
// This halves attention FLOPs vs sweeping all T keys, with zero error.
// ---------------------------------------------------------------------------
__global__ __launch_bounds__(32) void attn_wmma(
    const bf16_t* __restrict__ Q,    // [B,H,T,HS]
    const bf16_t* __restrict__ Ksrc, // [B,H,T,HS]
    const bf16_t* __restrict__ Vt,   // [B,H,HS,T]
    const float*  __restrict__ PS,   // [B,H,KT+1,HS] prefix sums of V
    bf16_t* __restrict__ Y) {        // [B,T,C]  (head-interleaved)
  __shared__ bf16_t Pl[16 * 32];     // P tile staging for C->A transpose

  const int lane = threadIdx.x & 31;
  const int cn = lane & 15, hi = lane >> 4;
  const int qtiles = Tc / 16;                 // 128
  const int bh = blockIdx.x / qtiles;
  const int qt = blockIdx.x % qtiles;
  const int b = bh / Hc, h = bh % Hc;

  const bf16_t* Qh = Q    + (size_t)bh * Tc * HSc;
  const bf16_t* Kh = Ksrc + (size_t)bh * Tc * HSc;
  const bf16_t* Vh = Vt   + (size_t)bh * HSc * Tc;
  const float*  psh = PS  + (size_t)bh * (KT + 1) * HSc;

  // Q tile (16 queries x 64 d) as two A fragments, loaded once.
  const v16bf aq0 = load_a16x32(Qh + (size_t)qt * 16 * HSc + 0,  HSc, lane);
  const v16bf aq1 = load_a16x32(Qh + (size_t)qt * 16 * HSc + 32, HSc, lane);

  float m8[8], l8[8];
  v8f O[4];
  #pragma unroll
  for (int r = 0; r < 8; ++r) { m8[r] = 0.f; l8[r] = 0.f; }
  #pragma unroll
  for (int f = 0; f < 4; ++f)
    #pragma unroll
    for (int r = 0; r < 8; ++r) O[f][r] = 0.f;

  const int qg_base = qt * 16 + 8 * hi;       // query row = qg_base + r
  const int steps = (qt * 16 + 15) / 32 + 1;  // causal 32-key tiles

  for (int kt = 0; kt < steps; ++kt) {
    const int jb = kt * 32;

    // ---- scores: two 16x16 tiles (32 keys), S = (Q K^T) * scale, masked->0
    v8f S[2];
    #pragma unroll
    for (int t2 = 0; t2 < 2; ++t2) {
      const int j0 = jb + 16 * t2;
      v16bf bk0 = load_b32x16T(Kh + (size_t)j0 * HSc + 0,  HSc, lane);
      v16bf bk1 = load_b32x16T(Kh + (size_t)j0 * HSc + 32, HSc, lane);
      v8f z;
      #pragma unroll
      for (int r = 0; r < 8; ++r) z[r] = 0.f;
      v8f s = wmma_bf16f32(aq0, bk0, z);
      s = wmma_bf16f32(aq1, bk1, s);
      const int jg = j0 + cn;                 // key column index
      #pragma unroll
      for (int r = 0; r < 8; ++r) {
        float sv = s[r] * ATT_SCALE;
        if (jg > qg_base + r) sv = 0.f;       // quirk: masked score == 0
        s[r] = sv;
      }
      S[t2] = s;
    }

    // ---- online softmax row statistics (reduce across 16 lanes per half)
    float fac[8];
    #pragma unroll
    for (int r = 0; r < 8; ++r) {
      float mx = fmaxf(S[0][r], S[1][r]);
      mx = fmaxf(mx, __shfl_xor(mx, 1));
      mx = fmaxf(mx, __shfl_xor(mx, 2));
      mx = fmaxf(mx, __shfl_xor(mx, 4));
      mx = fmaxf(mx, __shfl_xor(mx, 8));
      const float nm = fmaxf(m8[r], mx);
      fac[r] = __expf(m8[r] - nm);
      m8[r] = nm;
      S[0][r] = __expf(S[0][r] - nm);
      S[1][r] = __expf(S[1][r] - nm);
      float sm = S[0][r] + S[1][r];
      sm += __shfl_xor(sm, 1);
      sm += __shfl_xor(sm, 2);
      sm += __shfl_xor(sm, 4);
      sm += __shfl_xor(sm, 8);
      l8[r] = l8[r] * fac[r] + sm;
    }

    // ---- P (16x32) C-layout -> A-layout via LDS
    #pragma unroll
    for (int t2 = 0; t2 < 2; ++t2)
      #pragma unroll
      for (int r = 0; r < 8; ++r)
        Pl[(r + 8 * hi) * 32 + t2 * 16 + cn] = to_bf16(S[t2][r]);
    __syncthreads();                           // single-wave WG: cheap fence
    const v16bf ap = load_a16x32(Pl, 32, lane);

    // ---- O = O*fac + P @ V   (V columns are rows of Vt -> contiguous)
    #pragma unroll
    for (int f = 0; f < 4; ++f) {
      v16bf bv = load_b32x16T(Vh + (size_t)(16 * f) * Tc + jb, Tc, lane);
      #pragma unroll
      for (int r = 0; r < 8; ++r) O[f][r] *= fac[r];
      O[f] = wmma_bf16f32(ap, bv, O[f]);
    }
    __syncthreads();
  }

  // ---- analytic masked-tail correction (exact; see header comment)
  {
    const float tailN = (float)(Tc - 32 * steps);
    float em[8];
    #pragma unroll
    for (int r = 0; r < 8; ++r) {
      em[r] = __expf(-m8[r]);
      l8[r] += tailN * em[r];
    }
    #pragma unroll
    for (int f = 0; f < 4; ++f) {
      const int d = 16 * f + cn;
      const float tail_d = psh[(size_t)KT * HSc + d]
                         - psh[(size_t)steps * HSc + d];
      #pragma unroll
      for (int r = 0; r < 8; ++r) O[f][r] += em[r] * tail_d;
    }
  }

  // ---- normalize and write y in head-interleaved [B,T,C] bf16
  #pragma unroll
  for (int r = 0; r < 8; ++r) l8[r] = 1.f / l8[r];
  #pragma unroll
  for (int f = 0; f < 4; ++f)
    #pragma unroll
    for (int r = 0; r < 8; ++r) {
      const int tg = qt * 16 + r + 8 * hi;
      const float v = O[f][r] * l8[r];
      Y[((size_t)(b * Tc + tg)) * Cc + h * HSc + 16 * f + cn] = to_bf16(v);
    }
}

// ---------------------------------------------------------------------------
// Host-side orchestration.  Workspace layout (~48.5 MiB total):
//   xb 8M | Wqb 2M | Wkb 2M | Wvb 2M | Wpb 2M | Qb 8M | Kb 8M | Vtb 8M
//   | Yb 8M | PS 532K
// ---------------------------------------------------------------------------
extern "C" void kernel_launch(void* const* d_in, const int* in_sizes, int n_in,
                              void* d_out, int out_size, void* d_ws,
                              size_t ws_size, hipStream_t stream) {
  (void)in_sizes; (void)n_in; (void)out_size; (void)ws_size;

  const float* x  = (const float*)d_in[0];
  const float* Wq = (const float*)d_in[1];
  const float* Wk = (const float*)d_in[2];
  const float* Wv = (const float*)d_in[3];
  const float* Wp = (const float*)d_in[4];

  char* ws = (char*)d_ws;
  size_t off = 0;
  const size_t SZ_X  = (size_t)Mc * Cc * sizeof(bf16_t);   // 8 MiB
  const size_t SZ_W  = (size_t)Cc * Cc * sizeof(bf16_t);   // 2 MiB
  const size_t SZ_H  = (size_t)Bc * Hc * Tc * HSc * sizeof(bf16_t); // 8 MiB
  const size_t SZ_PS = (size_t)Bc * Hc * (KT + 1) * HSc * sizeof(float);

  bf16_t* xb  = (bf16_t*)(ws + off); off += SZ_X;
  bf16_t* Wqb = (bf16_t*)(ws + off); off += SZ_W;
  bf16_t* Wkb = (bf16_t*)(ws + off); off += SZ_W;
  bf16_t* Wvb = (bf16_t*)(ws + off); off += SZ_W;
  bf16_t* Wpb = (bf16_t*)(ws + off); off += SZ_W;
  bf16_t* Qb  = (bf16_t*)(ws + off); off += SZ_H;
  bf16_t* Kb  = (bf16_t*)(ws + off); off += SZ_H;
  bf16_t* Vtb = (bf16_t*)(ws + off); off += SZ_H;
  bf16_t* Yb  = (bf16_t*)(ws + off); off += SZ_X;
  float*  PS  = (float* )(ws + off); off += SZ_PS;

  // 1) convert inputs to bf16
  cvt_f32_bf16<<<2048, 256, 0, stream>>>(x,  xb,  Mc * Cc);
  cvt_f32_bf16<<<1024, 256, 0, stream>>>(Wq, Wqb, Cc * Cc);
  cvt_f32_bf16<<<1024, 256, 0, stream>>>(Wk, Wkb, Cc * Cc);
  cvt_f32_bf16<<<1024, 256, 0, stream>>>(Wv, Wvb, Cc * Cc);
  cvt_f32_bf16<<<1024, 256, 0, stream>>>(Wp, Wpb, Cc * Cc);

  // 2) QKV projections (waves: (Mc/32)*(Cc/64) = 2048; 4 waves/block)
  const int gemm_blocks = (Mc / 32) * (Cc / 64) / 4;      // 512
  gemm_bf16_wmma<<<gemm_blocks, 128, 0, stream>>>(xb, Wqb, (void*)Qb,  0);
  gemm_bf16_wmma<<<gemm_blocks, 128, 0, stream>>>(xb, Wkb, (void*)Kb,  0);
  gemm_bf16_wmma<<<gemm_blocks, 128, 0, stream>>>(xb, Wvb, (void*)Vtb, 1);

  // 3) V prefix sums (for the exact masked-tail fold)
  vprefix32<<<Bc * Hc, 64, 0, stream>>>(Vtb, PS);

  // 4) attention: one wave per (b, h, 16-query tile) -> 4096 waves
  attn_wmma<<<Bc * Hc * (Tc / 16), 32, 0, stream>>>(Qb, Kb, Vtb, PS, Yb);

  // 5) output projection -> f32 d_out
  gemm_bf16_wmma<<<gemm_blocks, 128, 0, stream>>>(Yb, Wpb, d_out, 2);
}
